// vAttn_71073118814768
// MI455X (gfx1250) — matrix operand order repaired and checked by reading, not compiled, not run on hardware
//
#include <hip/hip_runtime.h>
#include <hip/hip_bf16.h>
#include <stdint.h>

// ---------------------------------------------------------------------------
// bf16 WMMA types (CDNA5 / gfx1250, wave32)
// ---------------------------------------------------------------------------
typedef __attribute__((ext_vector_type(16))) __bf16         v16bf;
typedef __attribute__((ext_vector_type(8)))  float          v8f;
typedef __attribute__((ext_vector_type(4)))  unsigned int   u32x4;   // trivially constructible 128-bit

union BF16x16 {
  v16bf          v;
  unsigned short u[16];
  u32x4          q[2];     // two 16-byte halves -> b128 loads
};

__device__ __forceinline__ unsigned short f2bf(float f) {
  unsigned int u = __float_as_uint(f);
  unsigned int r = (u + 0x7FFFu + ((u >> 16) & 1u)) >> 16;   // RNE
  return (unsigned short)r;
}

// ---------------------------------------------------------------------------
// Elementwise f32 -> bf16
// ---------------------------------------------------------------------------
__global__ void cvt_f32_bf16(const float* __restrict__ in,
                             unsigned short* __restrict__ out, int n) {
  int i = blockIdx.x * blockDim.x + threadIdx.x;
  int stride = gridDim.x * blockDim.x;
  for (; i < n; i += stride) out[i] = f2bf(in[i]);
}

// Repack conv weights [Cout][Cin][3][3] f32 -> [tap=9][Cout][Cin] bf16
__global__ void pack_w3x3(const float* __restrict__ w,
                          unsigned short* __restrict__ wp, int Cout, int Cin) {
  int n = 9 * Cout * Cin;
  int i = blockIdx.x * blockDim.x + threadIdx.x;
  int stride = gridDim.x * blockDim.x;
  for (; i < n; i += stride) {
    int tap = i / (Cout * Cin);
    int rem = i - tap * (Cout * Cin);
    int co  = rem / Cin;
    int ci  = rem - co * Cin;
    int dy  = tap / 3, dx = tap - dy * 3;
    wp[i] = f2bf(w[((co * Cin + ci) * 3 + dy) * 3 + dx]);
  }
}

// ---------------------------------------------------------------------------
// conv3x3 (stride 1, SAME) as implicit GEMM on bf16 WMMA.
//   grid.x = B*64 (one (b, h) row per block), grid.y = Cout/16
//   block  = 128 threads = 4 waves; wave w owns the 16-wide w-tile w*16.
//   LDS stages 3 input rows x 64 channels TRANSPOSED: xs[row][w_pad][ci]
//   so each lane's B fragment (K=ci contiguous) is two ds_load_b128.
// mode 0: write bf16 out_bvc[b][t=w][v=h][co]  (attention-friendly layout)
// mode 1: write f32  out_f32[b][co][h][w] = xres + sigma * acc (residual)
// ---------------------------------------------------------------------------
__global__ __launch_bounds__(128)
void conv3x3_wmma(const unsigned short* __restrict__ xin,  // bf16 [B][Cin][64][64]
                  const unsigned short* __restrict__ wp,   // bf16 [9][Cout][Cin]
                  int Cin, int Cout, int mode,
                  unsigned short* __restrict__ out_bvc,
                  float* __restrict__ out_f32,
                  const float* __restrict__ xres,
                  const float* __restrict__ sigma_p)
{
  // [r 0..2][w_pad 0..65][ci 0..63]
  __shared__ __attribute__((aligned(16))) unsigned short xs[3 * 66 * 64];

  const int b    = blockIdx.x >> 6;
  const int h    = blockIdx.x & 63;
  const int co0  = blockIdx.y * 16;
  const int lane = threadIdx.x & 31;
  const int w0   = (threadIdx.x >> 5) * 16;
  const int hi   = lane >> 4;
  const int lo   = lane & 15;
  const int kbA  = hi ? 8 : 0;    // A fragment K split (0-7/16-23 vs 8-15/24-31)
  const int kbB  = hi ? 8 : 0;    // B fragment K split (0-7/16-23 vs 8-15/24-31)

  v8f acc = {};

  for (int cc = 0; cc < Cin; cc += 64) {
    // ---- stage 3 rows x 64 channels into LDS, transposed, zero-padded ----
    for (int j = threadIdx.x; j < 64 * 3; j += 128) {
      const int ci = j / 3, r = j - 3 * ci;
      const int hh = h + r - 1;
      // column pads (w = -1 and w = 64)
      xs[(r * 66 + 0)  * 64 + ci] = 0;
      xs[(r * 66 + 65) * 64 + ci] = 0;
      if ((unsigned)hh < 64u) {
        const u32x4* src4 = (const u32x4*)
            (xin + (((size_t)b * Cin + (cc + ci)) * 64 + hh) * 64);
        #pragma unroll
        for (int g = 0; g < 8; ++g) {             // 8 x 128-bit global reads
          u32x4 d = src4[g];
          #pragma unroll
          for (int q2 = 0; q2 < 4; ++q2) {
            unsigned int val = d[q2];
            int wbase = 1 + g * 8 + q2 * 2;
            xs[(r * 66 + wbase + 0) * 64 + ci] = (unsigned short)(val & 0xFFFFu);
            xs[(r * 66 + wbase + 1) * 64 + ci] = (unsigned short)(val >> 16);
          }
        }
      } else {
        #pragma unroll 4
        for (int wx = 1; wx < 65; ++wx) xs[(r * 66 + wx) * 64 + ci] = 0;
      }
    }
    __syncthreads();

    for (int kk = 0; kk < 2; ++kk) {      // two K=32 slices per 64-chan chunk
      const int ci0 = cc + kk * 32;
      for (int tap = 0; tap < 9; ++tap) {
        const int dy = tap / 3, dx = tap - dy * 3;

        // A = weight tile [16 x 32] -> two global_load_b128 per lane
        BF16x16 a;
        const unsigned short* wrow =
            wp + ((size_t)tap * Cout + (co0 + lo)) * Cin + ci0 + kbA;
        __builtin_prefetch(wrow + Cin, 0, 1);   // next Cout row -> global_prefetch
        a.q[0] = *(const u32x4*)(wrow);
        a.q[1] = *(const u32x4*)(wrow + 16);

        // B = activation tile [32 x 16] -> two ds_load_b128 per lane
        BF16x16 bm;
        const int wl = w0 + lo + dx;            // padded column index
        const unsigned short* bp =
            &xs[(unsigned)(dy * 66 + wl) * 64 + kk * 32 + kbB];
        bm.q[0] = *(const u32x4*)(bp);
        bm.q[1] = *(const u32x4*)(bp + 16);

        acc = __builtin_amdgcn_wmma_f32_16x16x32_bf16(
            false, a.v, false, bm.v, (short)0, acc, false, false);
      }
    }
    __syncthreads();
  }

  // ---- epilogue: C/D layout is VGPR r -> M = r + 8*hi, lane lo -> N ----
  const int t = w0 + lo;
  if (mode == 0) {
    // 8 consecutive co values -> pack into one 128-bit store
    u32x4 pk;
    #pragma unroll
    for (int p = 0; p < 4; ++p)
      pk[p] = (unsigned int)f2bf(acc[2 * p]) |
              ((unsigned int)f2bf(acc[2 * p + 1]) << 16);
    *(u32x4*)(&out_bvc[(((size_t)b * 64 + t) * 64 + h) * Cout + co0 + 8 * hi]) = pk;
  } else {
    const float sg = sigma_p[0];
    #pragma unroll
    for (int r = 0; r < 8; ++r) {
      int co = co0 + r + 8 * hi;
      size_t idx = (((size_t)b * Cout + co) * 64 + h) * 64 + t;
      out_f32[idx] = xres[idx] + sg * acc[r];
    }
  }
}

// ---------------------------------------------------------------------------
// Attention over V axis per (b,t): S = Q K^T (64x64, K=32), row softmax,
// AV = P V (64x128, K=64). One block per (b,t); 4 waves.
// V is staged TRANSPOSED in LDS (vs[c][w]) so AV B-fragments are contiguous.
// ---------------------------------------------------------------------------
__global__ __launch_bounds__(128)
void attn_wmma(const unsigned short* __restrict__ qb,   // bf16 [B][T][64][32]
               const unsigned short* __restrict__ kbuf, // bf16 [B][T][64][32]
               const unsigned short* __restrict__ vb,   // bf16 [B][T][64][128]
               unsigned short* __restrict__ avb,        // bf16 [B][128][64][64]
               float* __restrict__ attn_out)            // f32  [B*T][64][64]
{
  __shared__ __attribute__((aligned(16))) unsigned short qs[64 * 32];
  __shared__ __attribute__((aligned(16))) unsigned short ks[64 * 32];
  __shared__ __attribute__((aligned(16))) unsigned short vs[128 * 64]; // [c][w]
  __shared__ float                                       Sf[64 * 64];
  __shared__ __attribute__((aligned(16))) unsigned short Ps[64 * 64];

  const int bt   = blockIdx.x;
  const int b    = bt >> 6, t = bt & 63;
  const int lane = threadIdx.x & 31;
  const int mi   = threadIdx.x >> 5;      // wave id = M row-tile
  const int hi   = lane >> 4, lo = lane & 15;
  const int kbA  = hi ? 8 : 0;
  const int kbB  = hi ? 8 : 0;

  // ---- cooperative 128-bit loads into LDS ----
  {
    u32x4* qsv = (u32x4*)qs; const u32x4* qg = (const u32x4*)(qb   + (size_t)bt * 2048);
    u32x4* ksv = (u32x4*)ks; const u32x4* kg = (const u32x4*)(kbuf + (size_t)bt * 2048);
    for (int i = threadIdx.x; i < 256; i += 128) { qsv[i] = qg[i]; ksv[i] = kg[i]; }
    // V transpose: thread j handles row w = j>>1, channel half (j&1)*64
    {
      const int w     = threadIdx.x >> 1;
      const int chalf = (threadIdx.x & 1) * 64;
      const u32x4* src4 =
          (const u32x4*)(vb + (size_t)bt * 8192 + (size_t)w * 128 + chalf);
      #pragma unroll
      for (int g = 0; g < 8; ++g) {
        u32x4 d = src4[g];
        #pragma unroll
        for (int q2 = 0; q2 < 4; ++q2) {
          unsigned int val = d[q2];
          int c = chalf + g * 8 + q2 * 2;
          vs[(c + 0) * 64 + w] = (unsigned short)(val & 0xFFFFu);
          vs[(c + 1) * 64 + w] = (unsigned short)(val >> 16);
        }
      }
    }
  }
  __syncthreads();

  // ---- S = Q K^T : 4x4 tiles, one WMMA each (K = 32) ----
  for (int ni = 0; ni < 4; ++ni) {
    BF16x16 a, bm;
    const int row = mi * 16 + lo;
    const unsigned short* ap = &qs[row * 32 + kbA];
    a.q[0] = *(const u32x4*)(ap);
    a.q[1] = *(const u32x4*)(ap + 16);
    const int col = ni * 16 + lo;            // B[k][n] = K[n][k] -> contiguous
    const unsigned short* bp = &ks[col * 32 + kbB];
    bm.q[0] = *(const u32x4*)(bp);
    bm.q[1] = *(const u32x4*)(bp + 16);

    v8f s = {};
    s = __builtin_amdgcn_wmma_f32_16x16x32_bf16(
        false, a.v, false, bm.v, (short)0, s, false, false);
    #pragma unroll
    for (int r = 0; r < 8; ++r)
      Sf[(mi * 16 + r + 8 * hi) * 64 + ni * 16 + lo] = s[r];
  }
  __syncthreads();

  // ---- row softmax in f32; emit attn output + bf16 P ----
  if (threadIdx.x < 64) {
    const int row = threadIdx.x;
    float m = -1e30f;
    for (int j = 0; j < 64; ++j) m = fmaxf(m, Sf[row * 64 + j]);
    float ssum = 0.f;
    for (int j = 0; j < 64; ++j) {
      float e = __expf(Sf[row * 64 + j] - m);
      ssum += e;
      Sf[row * 64 + j] = e;
    }
    const float inv = 1.f / ssum;
    for (int j = 0; j < 64; ++j) {
      float p = Sf[row * 64 + j] * inv;
      Ps[row * 64 + j] = f2bf(p);
      attn_out[(size_t)bt * 4096 + row * 64 + j] = p;
    }
  }
  __syncthreads();

  // ---- AV = P V : M=64, N=128, K=64 (2 WMMA K-steps per tile) ----
  for (int ni = 0; ni < 8; ++ni) {
    v8f o = {};
    for (int ks2 = 0; ks2 < 2; ++ks2) {
      BF16x16 a, bm;
      const int row = mi * 16 + lo;
      const unsigned short* ap = &Ps[row * 64 + ks2 * 32 + kbA];
      a.q[0] = *(const u32x4*)(ap);
      a.q[1] = *(const u32x4*)(ap + 16);
      // B[k=w][n=c] = V[w][c] = vs[c][w] -> contiguous in transposed layout
      const unsigned short* bp = &vs[(ni * 16 + lo) * 64 + ks2 * 32 + kbB];
      bm.q[0] = *(const u32x4*)(bp);
      bm.q[1] = *(const u32x4*)(bp + 16);
      o = __builtin_amdgcn_wmma_f32_16x16x32_bf16(
          false, a.v, false, bm.v, (short)0, o, false, false);
    }
    #pragma unroll
    for (int r = 0; r < 8; ++r) {
      int c = ni * 16 + lo;
      int v = mi * 16 + r + 8 * hi;
      avb[(((size_t)b * 128 + c) * 64 + v) * 64 + t] = f2bf(o[r]);
    }
  }
}

// ---------------------------------------------------------------------------
// Host launcher
// ---------------------------------------------------------------------------
extern "C" void kernel_launch(void* const* d_in, const int* in_sizes, int n_in,
                              void* d_out, int out_size, void* d_ws, size_t ws_size,
                              hipStream_t stream) {
  (void)in_sizes; (void)n_in; (void)out_size; (void)ws_size;

  const float* x     = (const float*)d_in[0];  // [16][256][64][64]
  const float* wq    = (const float*)d_in[1];  // [32][256][3][3]
  const float* wk    = (const float*)d_in[2];
  const float* wv    = (const float*)d_in[3];  // [128][256][3][3]
  const float* wo    = (const float*)d_in[4];  // [256][128][3][3]
  const float* sigma = (const float*)d_in[5];

  char* ws = (char*)d_ws;
  size_t off = 0;
  auto alloc = [&](size_t bytes) -> void* {
    void* p = ws + off;
    off = (off + bytes + 255) & ~(size_t)255;
    return p;
  };
  unsigned short* xb    = (unsigned short*)alloc((size_t)16 * 256 * 64 * 64 * 2);
  unsigned short* qbuf  = (unsigned short*)alloc((size_t)16 * 64 * 64 * 32  * 2);
  unsigned short* kbuf  = (unsigned short*)alloc((size_t)16 * 64 * 64 * 32  * 2);
  unsigned short* vbuf  = (unsigned short*)alloc((size_t)16 * 64 * 64 * 128 * 2);
  unsigned short* avbuf = (unsigned short*)alloc((size_t)16 * 128 * 64 * 64 * 2);
  unsigned short* wqp   = (unsigned short*)alloc((size_t)9 * 32  * 256 * 2);
  unsigned short* wkp   = (unsigned short*)alloc((size_t)9 * 32  * 256 * 2);
  unsigned short* wvp   = (unsigned short*)alloc((size_t)9 * 128 * 256 * 2);
  unsigned short* wop   = (unsigned short*)alloc((size_t)9 * 256 * 128 * 2);

  float* out_f32  = (float*)d_out;                       // [16][256][64][64]
  float* attn_out = (float*)d_out + (size_t)16 * 256 * 64 * 64;

  // 1) precision conversion + weight repack
  cvt_f32_bf16<<<4096, 256, 0, stream>>>(x, xb, 16 * 256 * 64 * 64);
  pack_w3x3<<<(9 * 32  * 256 + 255) / 256, 256, 0, stream>>>(wq, wqp, 32, 256);
  pack_w3x3<<<(9 * 32  * 256 + 255) / 256, 256, 0, stream>>>(wk, wkp, 32, 256);
  pack_w3x3<<<(9 * 128 * 256 + 255) / 256, 256, 0, stream>>>(wv, wvp, 128, 256);
  pack_w3x3<<<(9 * 256 * 128 + 255) / 256, 256, 0, stream>>>(wo, wop, 256, 128);

  // 2) q/k/v conv3x3 as WMMA implicit GEMM (outputs in [B][T][V][C] bf16)
  conv3x3_wmma<<<dim3(1024, 2),  128, 0, stream>>>(xb, wqp, 256, 32, 0,
                                                   qbuf, nullptr, nullptr, nullptr);
  conv3x3_wmma<<<dim3(1024, 2),  128, 0, stream>>>(xb, wkp, 256, 32, 0,
                                                   kbuf, nullptr, nullptr, nullptr);
  conv3x3_wmma<<<dim3(1024, 8),  128, 0, stream>>>(xb, wvp, 256, 128, 0,
                                                   vbuf, nullptr, nullptr, nullptr);

  // 3) batched attention over the V axis (also writes attn output)
  attn_wmma<<<1024, 128, 0, stream>>>(qbuf, kbuf, vbuf, avbuf, attn_out);

  // 4) output conv3x3 + residual epilogue (f32 out)
  conv3x3_wmma<<<dim3(1024, 16), 128, 0, stream>>>(avbuf, wop, 128, 256, 1,
                                                   nullptr, out_f32, x, sigma);
}
